// AutoRegressive_88837103551116
// MI455X (gfx1250) — compile-verified, matrix operand order checked by reading, not run on hardware
//
#include <hip/hip_runtime.h>
#include <hip/hip_bf16.h>

typedef __attribute__((ext_vector_type(16))) _Float16 v16h;
typedef __attribute__((ext_vector_type(8)))  _Float16 v8h;
typedef __attribute__((ext_vector_type(8)))  float    v8f;

#define HID     64
#define GATES   256   // 4*HID
#define BT      16    // batch rows per workgroup
#define NIN     16    // layer-0 input width (F + C)
#define NIN_PAD 32    // K padded to one WMMA k-step
#define NTHREADS 128  // 4 waves of 32

// LDS layout (bytes):
//   sWhh0 [256][64] f16 : 32768   (staging for reg load)
//   sWih1 [256][64] f16 : 32768
//   sWhh1 [256][64] f16 : 32768
//   sWih0 [256][32] f16 : 16384   (cols >= 16 zero-padded)
//   sH0   [16][64]  f16 :  2048
//   sH1   [16][64]  f16 :  2048
//   sB0   [256]     f32 :  1024
//   sB1   [256]     f32 :  1024
//   sWd   [8][64]   f32 :  2048
//   sBd   [8]       f32 :    32
//   sElem [16][8]   f32 :   512
//   sLenX [16]      i32 :    64
//   sLenC [16]      i32 :    64
constexpr size_t SMEM_BYTES =
    (size_t)GATES * HID * 2 * 3 +
    (size_t)GATES * NIN_PAD * 2 +
    (size_t)BT * HID * 2 * 2 +
    (size_t)GATES * 4 * 2 +
    (size_t)8 * HID * 4 + 8 * 4 +
    (size_t)BT * 8 * 4 +
    (size_t)BT * 4 * 2;   // = 123552

__device__ __forceinline__ v8f wmma16(v16h a, v16h b, v8f c) {
  // D = A(16x32 f16) * B(32x16 f16) + C(16x16 f32); D does not clobber C.
  return __builtin_amdgcn_wmma_f32_16x16x32_f16(false, a, false, b, (short)0, c,
                                                false, false);
}

// Load a 16x32 f16 tile from an LDS matrix (row stride `ld` halves) at
// (row0, k0), per the CDNA5 16-bit A/B operand striping:
//   lane = {m, g}:  halves 0..7  -> K = g*8 + 0..7
//                   halves 8..15 -> K = 16 + g*8 + 0..7
__device__ __forceinline__ v16h ld_tile(const _Float16* base, int ld, int row0,
                                        int k0, int lane) {
  const _Float16* p =
      base + (size_t)(row0 + (lane & 15)) * ld + k0 + ((lane >> 4) << 3);
  v8h lo = *(const v8h*)p;
  v8h hi = *(const v8h*)(p + 16);
  v16h r;
#pragma unroll
  for (int i = 0; i < 8; ++i) { r[i] = lo[i]; r[i + 8] = hi[i]; }
  return r;
}

__device__ __forceinline__ float sigmoidf_fast(float x) {
  x = fminf(fmaxf(x, -30.f), 30.f);
  return 1.f / (1.f + __expf(-x));
}
__device__ __forceinline__ float tanhf_fast(float x) {
  x = fminf(fmaxf(x, -15.f), 15.f);
  float e = __expf(2.f * x);
  return (e - 1.f) / (e + 1.f);
}

// Build the layer-0 input A-tile (16x32, upper K half zero) straight from
// global x[b0+m][t][0:16] — per-lane data is 8 contiguous f32.
__device__ __forceinline__ v16h make_x_enc(const float* __restrict__ x, int b0,
                                           int TW, int t, int lane) {
  int m = lane & 15, g = lane >> 4;
  const float* row = x + ((size_t)(b0 + m) * TW + t) * NIN + g * 8;
  v16h r;
#pragma unroll
  for (int j = 0; j < 8; ++j) {
    r[j] = (_Float16)row[j];
    r[j + 8] = (_Float16)0.f;
  }
  return r;
}

// Decoder input A-tile: cols 0..7 = element (LDS), 8..15 = context (global).
__device__ __forceinline__ v16h make_x_dec(const float* sElem,
                                           const float* __restrict__ context,
                                           int b0, int TC, int t, int lane) {
  int m = lane & 15, g = lane >> 4;
  v16h r;
#pragma unroll
  for (int j = 0; j < 8; ++j) {
    int k = g * 8 + j;
    float v = (k < 8) ? sElem[m * 8 + k]
                      : context[((size_t)(b0 + m) * TC + t) * 8 + (k - 8)];
    r[j] = (_Float16)v;
    r[j + 8] = (_Float16)0.f;
  }
  return r;
}

// LSTM cell update on this wave's 16x16 tile; freeze where !(tcmp < len8[r]).
__device__ __forceinline__ void lstm_update(const v8f* g4, v8f& creg, v8f& hreg,
                                            const int* len8, int tcmp) {
#pragma unroll
  for (int r = 0; r < 8; ++r) {
    float iv = sigmoidf_fast(g4[0][r]);
    float fv = sigmoidf_fast(g4[1][r]);
    float gv = tanhf_fast(g4[2][r]);
    float ov = sigmoidf_fast(g4[3][r]);
    float cn = fv * creg[r] + iv * gv;
    float hn = ov * tanhf_fast(cn);
    bool live = tcmp < len8[r];
    creg[r] = live ? cn : creg[r];
    hreg[r] = live ? hn : hreg[r];
  }
}

__device__ __forceinline__ void store_h(_Float16* sH, const v8f& hreg, int ht,
                                        int lane) {
  int col = ht * 16 + (lane & 15);
#pragma unroll
  for (int r = 0; r < 8; ++r) {
    int m = r + ((lane >> 4) << 3);
    sH[m * HID + col] = (_Float16)hreg[r];
  }
}

__global__ __launch_bounds__(NTHREADS)
__attribute__((amdgpu_waves_per_eu(1))) void lstm_seq2seq_kernel(
    const float* __restrict__ x, const int* __restrict__ lengths_x,
    const float* __restrict__ context, const int* __restrict__ lengths_ctx,
    const float* __restrict__ Wih0, const float* __restrict__ Whh0,
    const float* __restrict__ bih0, const float* __restrict__ bhh0,
    const float* __restrict__ Wih1, const float* __restrict__ Whh1,
    const float* __restrict__ bih1, const float* __restrict__ bhh1,
    const float* __restrict__ Wd, const float* __restrict__ bd,
    float* __restrict__ out, int B, int TW, int TC) {
  extern __shared__ char smem[];
  _Float16* sWhh0 = (_Float16*)smem;
  _Float16* sWih1 = sWhh0 + GATES * HID;
  _Float16* sWhh1 = sWih1 + GATES * HID;
  _Float16* sWih0 = sWhh1 + GATES * HID;
  _Float16* sH0   = sWih0 + GATES * NIN_PAD;
  _Float16* sH1   = sH0 + BT * HID;
  float* sB0   = (float*)(sH1 + BT * HID);
  float* sB1   = sB0 + GATES;
  float* sWd   = sB1 + GATES;
  float* sBd   = sWd + 8 * HID;
  float* sElem = sBd + 8;
  int* sLenX   = (int*)(sElem + BT * 8);
  int* sLenC   = sLenX + BT;

  const int tid  = threadIdx.x;
  const int lane = tid & 31;
  const int ht   = tid >> 5;           // wave id == hidden-column tile
  const int b0   = blockIdx.x * BT;

  // ---- one-time: weights/biases -> LDS (f32 -> f16 staging) ----
  for (int i = tid; i < GATES * HID; i += NTHREADS) {
    sWhh0[i] = (_Float16)Whh0[i];
    sWih1[i] = (_Float16)Wih1[i];
    sWhh1[i] = (_Float16)Whh1[i];
  }
  for (int i = tid; i < GATES * NIN_PAD; i += NTHREADS) {
    int n = i >> 5, k = i & (NIN_PAD - 1);
    sWih0[i] = (k < NIN) ? (_Float16)Wih0[n * NIN + k] : (_Float16)0.f;
  }
  for (int i = tid; i < GATES; i += NTHREADS) {
    sB0[i] = bih0[i] + bhh0[i];
    sB1[i] = bih1[i] + bhh1[i];
  }
  for (int i = tid; i < 8 * HID; i += NTHREADS) sWd[i] = Wd[i];
  if (tid < 8) sBd[tid] = bd[tid];
  if (tid < BT) {
    sLenX[tid] = lengths_x[b0 + tid];
    sLenC[tid] = lengths_ctx[b0 + tid];
  }
  for (int i = tid; i < BT * HID; i += NTHREADS) {
    sH0[i] = (_Float16)0.f;
    sH1[i] = (_Float16)0.f;
  }
  __syncthreads();

  // ---- hoist loop-invariant B-operands + bias C-tiles into registers ----
  v16h wih0r[4];
  v16h whh0r[4][2], wih1r[4][2], whh1r[4][2];
  v8f  b0r[4], b1r[4];
#pragma unroll
  for (int g = 0; g < 4; ++g) {
    int n0 = g * HID + ht * 16;
    wih0r[g]    = ld_tile(sWih0, NIN_PAD, n0, 0, lane);
    whh0r[g][0] = ld_tile(sWhh0, HID, n0, 0, lane);
    whh0r[g][1] = ld_tile(sWhh0, HID, n0, 32, lane);
    wih1r[g][0] = ld_tile(sWih1, HID, n0, 0, lane);
    wih1r[g][1] = ld_tile(sWih1, HID, n0, 32, lane);
    whh1r[g][0] = ld_tile(sWhh1, HID, n0, 0, lane);
    whh1r[g][1] = ld_tile(sWhh1, HID, n0, 32, lane);
    float bv0 = sB0[n0 + (lane & 15)];
    float bv1 = sB1[n0 + (lane & 15)];
#pragma unroll
    for (int r = 0; r < 8; ++r) {
      b0r[g][r] = bv0;
      b1r[g][r] = bv1;
    }
  }
  // per-lane length registers (row m = r + 8*(lane>>4))
  int lenx8[8], lenc8[8];
#pragma unroll
  for (int r = 0; r < 8; ++r) {
    int m = r + ((lane >> 4) << 3);
    lenx8[r] = sLenX[m];
    lenc8[r] = sLenC[m];
  }

  v8f c0reg = {0.f, 0.f, 0.f, 0.f, 0.f, 0.f, 0.f, 0.f};
  v8f h0reg = c0reg, c1reg = c0reg, h1reg = c0reg;

  // one full 2-layer LSTM step; 4 barriers, all weights register-resident
  auto lstm_step = [&](v16h ax, const int* len8, int tcmp) {
    v8f g4[4];
    v16h a0 = ld_tile(sH0, HID, 0, 0, lane);
    v16h a1 = ld_tile(sH0, HID, 0, 32, lane);
#pragma unroll
    for (int g = 0; g < 4; ++g) {
      v8f acc = wmma16(ax, wih0r[g], b0r[g]);   // bias as C, preserved
      acc = wmma16(a0, whh0r[g][0], acc);
      acc = wmma16(a1, whh0r[g][1], acc);
      g4[g] = acc;
    }
    __syncthreads();                             // all reads of sH0 done
    lstm_update(g4, c0reg, h0reg, len8, tcmp);
    store_h(sH0, h0reg, ht, lane);
    __syncthreads();                             // new h0 visible

    v16h d0 = ld_tile(sH0, HID, 0, 0, lane);
    v16h d1 = ld_tile(sH0, HID, 0, 32, lane);
    v16h r0 = ld_tile(sH1, HID, 0, 0, lane);
    v16h r1 = ld_tile(sH1, HID, 0, 32, lane);
#pragma unroll
    for (int g = 0; g < 4; ++g) {
      v8f acc = wmma16(d0, wih1r[g][0], b1r[g]);
      acc = wmma16(d1, wih1r[g][1], acc);
      acc = wmma16(r0, whh1r[g][0], acc);
      acc = wmma16(r1, whh1r[g][1], acc);
      g4[g] = acc;
    }
    __syncthreads();                             // all reads of sH1 done
    lstm_update(g4, c1reg, h1reg, len8, tcmp);
    store_h(sH1, h1reg, ht, lane);
    __syncthreads();                             // new h1 visible
  };

  // =================== encoder scan ===================
  for (int t = 0; t < TW; ++t) {
    lstm_step(make_x_enc(x, b0, TW, t, lane), lenx8, t);
  }

  // ---- element = hT[layer1] @ Wd^T + bd ; emit out[:,0,:] ----
  {
    int m = tid >> 3, f = tid & 7;
    float a = sBd[f];
#pragma unroll
    for (int k = 0; k < HID; ++k) a += (float)sH1[m * HID + k] * sWd[f * HID + k];
    sElem[m * 8 + f] = a;
    out[((size_t)(b0 + m) * TC) * 8 + f] = (0 < sLenC[m]) ? a : -999.f;
  }
  __syncthreads();

  // =================== decoder scan ===================
  for (int t = 0; t < TC - 1; ++t) {
    lstm_step(make_x_dec(sElem, context, b0, TC, t, lane), lenc8, t + 1);
    // out[:, t+1, :] = h1 @ Wd^T + bd (or -999 past length). The read of sH1
    // here is separated from its next write by >= 2 barriers inside lstm_step.
    int m = tid >> 3, f = tid & 7;
    float a = sBd[f];
#pragma unroll
    for (int k = 0; k < HID; ++k) a += (float)sH1[m * HID + k] * sWd[f * HID + k];
    out[((size_t)(b0 + m) * TC + (t + 1)) * 8 + f] =
        ((t + 1) < sLenC[m]) ? a : -999.f;
  }
}

extern "C" void kernel_launch(void* const* d_in, const int* in_sizes, int n_in,
                              void* d_out, int out_size, void* d_ws, size_t ws_size,
                              hipStream_t stream) {
  const float* x           = (const float*)d_in[0];
  const int*   lengths_x   = (const int*)d_in[1];
  const float* context     = (const float*)d_in[2];
  const int*   lengths_ctx = (const int*)d_in[3];
  const float* Wih0 = (const float*)d_in[4];
  const float* Whh0 = (const float*)d_in[5];
  const float* bih0 = (const float*)d_in[6];
  const float* bhh0 = (const float*)d_in[7];
  const float* Wih1 = (const float*)d_in[8];
  const float* Whh1 = (const float*)d_in[9];
  const float* bih1 = (const float*)d_in[10];
  const float* bhh1 = (const float*)d_in[11];
  const float* Wd   = (const float*)d_in[12];
  const float* bd   = (const float*)d_in[13];
  float* out = (float*)d_out;

  int B  = in_sizes[1];                 // lengths_x count
  int TW = in_sizes[0] / (B * NIN);     // x = [B, TW, 16]
  int TC = in_sizes[2] / (B * 8);       // context = [B, TC, 8]

  dim3 grid(B / BT), block(NTHREADS);
  lstm_seq2seq_kernel<<<grid, block, SMEM_BYTES, stream>>>(
      x, lengths_x, context, lengths_ctx, Wih0, Whh0, bih0, bhh0, Wih1, Whh1,
      bih1, bhh1, Wd, bd, out, B, TW, TC);
}
